// GNN7_19378892439967
// MI455X (gfx1250) — matrix operand order repaired and checked by reading, not compiled, not run on hardware
//
#include <hip/hip_runtime.h>
#include <cstdint>

// ---------------- problem constants (from reference) ----------------
#define NN      50000
#define EE      400000    // multiple of 16 -> edge kernel needs no tail guard
#define BB      64
#define IN_DIM  512
#define MID_DIM 128
#define ED_DIM  64
#define HC      256      // H * C
#define CC      128      // per-head channels

// ---------------- WMMA types / helpers ----------------
typedef __attribute__((ext_vector_type(16))) __bf16    v16bf;
typedef __attribute__((ext_vector_type(8)))  float     v8f;
typedef __attribute__((ext_vector_type(4)))  unsigned  v4u;

union BF16x16 { v16bf v; v4u q[2]; };
union F32x8   { v8f   v; float e[8]; };

// convert two fp32 to packed bf16 (round-to-nearest-even)
__device__ inline unsigned pack2bf(float lo, float hi) {
  unsigned a = __builtin_bit_cast(unsigned, lo);
  a += 0x7FFFu + ((a >> 16) & 1u);
  unsigned b = __builtin_bit_cast(unsigned, hi);
  b += 0x7FFFu + ((b >> 16) & 1u);
  return (a >> 16) | (b & 0xFFFF0000u);
}

__device__ inline v16bf ldsFrag(const unsigned* p) {   // 32B contiguous -> 2x ds_load_b128
  BF16x16 f;
  f.q[0] = *(const v4u*)(p);
  f.q[1] = *(const v4u*)(p + 4);
  return f.v;
}

__device__ inline void atomicMaxF(float* a, float v) {
  int* ai = (int*)a;
  int cur = __float_as_int(*a);
  while (v > __int_as_float(cur)) {
    int old = atomicCAS(ai, cur, __float_as_int(v));
    if (old == cur) break;
    cur = old;
  }
}

// ---------------- kernel 1: Y[M,256] = X[M,K] @ W[K,256], bf16 WMMA ----------------
// block = 128 threads (4 waves); tile 64x64; LDS tiles held in WMMA fragment order.
// Block-uniform full/partial split keeps the hot path free of per-lane exec masking.
__global__ __launch_bounds__(128)
void gemm_xw_wmma(const float* __restrict__ X, const float* __restrict__ W,
                  float* __restrict__ Y, int M, int K) {
  __shared__ __align__(16) unsigned AfS[4][32][8];   // [wave-tile][lane][dword] 4 KB
  __shared__ __align__(16) unsigned BfS[4][32][8];   // [n-subtile][lane][dword] 4 KB
  const int t = threadIdx.x, wv = t >> 5, lane = t & 31;
  const int m0 = blockIdx.x * 64, n0 = blockIdx.y * 64;
  const bool full = (m0 + 64 <= M);                  // uniform across the block

  v8f acc[4] = {};
  for (int k0 = 0; k0 < K; k0 += 32) {
    __syncthreads();
    // A tile 64x32 -> fragment layout. 1024 bf16-pairs, 8 per thread.
    if (full) {
#pragma unroll
      for (int j = 0; j < 8; ++j) {
        int p2 = j * 128 + t;
        int r  = p2 >> 4;              // 16 pairs per 32-wide row
        int c  = (p2 & 15) * 2;
        const float* src = X + (size_t)(m0 + r) * K + k0 + c;
        int ln   = (r & 15) + ((c >> 3) & 1) * 16;
        int slot = (c & 7) + ((c >> 4) & 1) * 8;
        AfS[r >> 4][ln][slot >> 1] = pack2bf(src[0], src[1]);
      }
    } else {
#pragma unroll
      for (int j = 0; j < 8; ++j) {
        int p2 = j * 128 + t;
        int r  = p2 >> 4;
        int c  = (p2 & 15) * 2;
        float lo = 0.f, hi = 0.f;
        if (m0 + r < M) {
          const float* src = X + (size_t)(m0 + r) * K + k0 + c;
          lo = src[0]; hi = src[1];
        }
        int ln   = (r & 15) + ((c >> 3) & 1) * 16;
        int slot = (c & 7) + ((c >> 4) & 1) * 8;
        AfS[r >> 4][ln][slot >> 1] = pack2bf(lo, hi);
      }
    }
    // B tile 32x64 -> fragment layout; pairs along k (two coalesced row reads).
#pragma unroll
    for (int j = 0; j < 8; ++j) {
      int p2 = j * 128 + t;
      int k  = (p2 >> 6) * 2;          // 0,2,..,30
      int c  = p2 & 63;
      const float* src = W + (size_t)(k0 + k) * HC + n0 + c;
      int ln = (c & 15) + ((k >> 3) & 1) * 16;
      BfS[c >> 4][ln][((k & 7) + ((k >> 4) & 1) * 8) >> 1] = pack2bf(src[0], src[HC]);
    }
    __syncthreads();

    BF16x16 a; a.v = ldsFrag(&AfS[wv][lane][0]);
#pragma unroll
    for (int nt = 0; nt < 4; ++nt) {
      v16bf b = ldsFrag(&BfS[nt][lane][0]);
      acc[nt] = __builtin_amdgcn_wmma_f32_16x16x32_bf16(
          false, a.v, false, b, (short)0, acc[nt], false, false);
    }
  }

  if (full) {
#pragma unroll
    for (int nt = 0; nt < 4; ++nt) {
      F32x8 r; r.v = acc[nt];
#pragma unroll
      for (int q = 0; q < 8; ++q) {
        int ml  = q + ((lane & 16) ? 8 : 0);
        int row = m0 + wv * 16 + ml;
        Y[(size_t)row * HC + n0 + nt * 16 + (lane & 15)] = r.e[q];
      }
    }
  } else {
#pragma unroll
    for (int nt = 0; nt < 4; ++nt) {
      F32x8 r; r.v = acc[nt];
#pragma unroll
      for (int q = 0; q < 8; ++q) {
        int ml  = q + ((lane & 16) ? 8 : 0);
        int row = m0 + wv * 16 + ml;
        if (row < M) Y[(size_t)row * HC + n0 + nt * 16 + (lane & 15)] = r.e[q];
      }
    }
  }
}

// ---------------- kernel 2: fused edge projection (WMMA in LDS) + GATv2 logits ----
// E is a multiple of the 16-edge tile, so staging needs no bounds guards at all.
__global__ __launch_bounds__(128)
void edge_logits_wmma(const float* __restrict__ edge_attr,  // [E,64]
                      const float* __restrict__ we,         // [64,256]
                      const float* __restrict__ xl,         // [N,256]
                      const float* __restrict__ xr,         // [N,256]
                      const float* __restrict__ att,        // [2,128]
                      const int*   __restrict__ edge_index, // [2,E]
                      float* __restrict__ logits,           // [E,2]
                      float* __restrict__ mx) {             // [N,2]
  __shared__ __align__(16) unsigned EaF[2][32][8];       // A frags, 2 k-steps (2 KB)
  __shared__ __align__(16) unsigned WeF[2][16][32][8];   // B frags [ks][ct][lane] (32 KB)
  __shared__ float EpS[16][256];                         // projection, LDS-only (16 KB)
  __shared__ float attS[256];
  const int t = threadIdx.x, wv = t >> 5, lane = t & 31;
  const long e0 = (long)blockIdx.x * 16;

  // stage edge_attr 16x64 -> A fragments (512 pairs, 4 per thread, no guard)
#pragma unroll
  for (int j = 0; j < 4; ++j) {
    int p2 = j * 128 + t;
    int r  = p2 >> 5;
    int c  = (p2 & 31) * 2;
    const float* src = edge_attr + (size_t)(e0 + r) * ED_DIM + c;
    int kk = c & 31;
    int ln   = (r & 15) + ((kk >> 3) & 1) * 16;
    int slot = (kk & 7) + ((kk >> 4) & 1) * 8;
    EaF[c >> 5][ln][slot >> 1] = pack2bf(src[0], src[1]);
  }
  // stage we 64x256 -> B fragments; pairs along k (8192 pairs, 64 per thread)
  for (int j = 0; j < 64; ++j) {
    int p2 = j * 128 + t;
    int k  = (p2 >> 8) * 2;          // 0,2,..,62
    int c  = p2 & 255;
    const float* src = we + (size_t)k * HC + c;
    int kk = k & 31;
    int ln = (c & 15) + ((kk >> 3) & 1) * 16;
    WeF[k >> 5][c >> 4][ln][((kk & 7) + ((kk >> 4) & 1) * 8) >> 1] = pack2bf(src[0], src[HC]);
  }
  attS[t] = att[t];
  attS[t + 128] = att[t + 128];
  __syncthreads();

  // Phase A: wave wv computes cols 64*wv..64*wv+63 of the 16x256 projection.
  v8f acc[4] = {};
#pragma unroll
  for (int ks = 0; ks < 2; ++ks) {
    BF16x16 a; a.v = ldsFrag(&EaF[ks][lane][0]);
#pragma unroll
    for (int nt = 0; nt < 4; ++nt) {
      v16bf b = ldsFrag(&WeF[ks][wv * 4 + nt][lane][0]);
      acc[nt] = __builtin_amdgcn_wmma_f32_16x16x32_bf16(
          false, a.v, false, b, (short)0, acc[nt], false, false);
    }
  }
#pragma unroll
  for (int nt = 0; nt < 4; ++nt) {
    F32x8 r; r.v = acc[nt];
#pragma unroll
    for (int q = 0; q < 8; ++q) {
      int ml = q + ((lane & 16) ? 8 : 0);
      EpS[ml][wv * 64 + nt * 16 + (lane & 15)] = r.e[q];
    }
  }
  __syncthreads();

  // Phase B: 32 (edge,head) tasks, 8 per wave; each lane covers 4 channels.
#pragma unroll
  for (int j = 0; j < 8; ++j) {
    int task = wv * 8 + j;
    int el = task >> 1, hh = task & 1;
    long e = e0 + el;
    int src = edge_index[e];
    int dst = edge_index[(size_t)EE + e];
    float partial = 0.f;
#pragma unroll
    for (int jj = 0; jj < 4; ++jj) {
      int c  = lane + 32 * jj;
      int hc = hh * CC + c;
      float s = xl[(size_t)src * HC + hc] + xr[(size_t)dst * HC + hc] + EpS[el][hc];
      float m = s > 0.f ? s : 0.2f * s;        // leaky_relu(., 0.2)
      partial += m * attS[hc];
    }
#pragma unroll
    for (int off = 16; off > 0; off >>= 1)
      partial += __shfl_xor(partial, off, 32);
    if (lane == 0) {
      logits[(size_t)e * 2 + hh] = partial;
      atomicMaxF(&mx[(size_t)dst * 2 + hh], partial);
    }
  }
}

// ---------------- kernel 3: ea = exp(logit - mx[dst]); denom += ea ----------------
__global__ void edge_softmax_denom(const float* __restrict__ logits,
                                   const int* __restrict__ edge_index,
                                   const float* __restrict__ mx,
                                   float* __restrict__ eav, float* __restrict__ denom) {
  long i = (long)blockIdx.x * blockDim.x + threadIdx.x;
  if (i >= (long)EE * 2) return;
  long e = i >> 1; int hh = (int)(i & 1);
  int dst = edge_index[(size_t)EE + e];
  float v = __expf(logits[i] - mx[(size_t)dst * 2 + hh]);
  eav[i] = v;
  atomicAdd(&denom[(size_t)dst * 2 + hh], v);
}

// ---------------- kernel 4: agg[dst] += xl[src] * alpha (wave per edge) ----------
__global__ __launch_bounds__(256)
void edge_aggregate(const float* __restrict__ xl, const float* __restrict__ eav,
                    const float* __restrict__ denom, const int* __restrict__ edge_index,
                    float* __restrict__ agg) {
  const int lane = threadIdx.x & 31;
  const long e = (long)blockIdx.x * 8 + (threadIdx.x >> 5);
  int src = edge_index[e];
  int dst = edge_index[(size_t)EE + e];
  float a0 = eav[e * 2 + 0] / (denom[(size_t)dst * 2 + 0] + 1e-16f);
  float a1 = eav[e * 2 + 1] / (denom[(size_t)dst * 2 + 1] + 1e-16f);
#pragma unroll
  for (int j = 0; j < 8; ++j) {
    int hc = lane + 32 * j;
    float alpha = (hc < CC) ? a0 : a1;
    atomicAdd(&agg[(size_t)dst * HC + hc], xl[(size_t)src * HC + hc] * alpha);
  }
}

// ---------------- kernel 5: mean over heads + bias (+relu) ----------------
__global__ void node_finalize(const float* __restrict__ agg, const float* __restrict__ bias,
                              float* __restrict__ out, int relu) {
  long i = (long)blockIdx.x * blockDim.x + threadIdx.x;
  if (i >= (long)NN * CC) return;
  long n = i / CC; int c = (int)(i % CC);
  float v = 0.5f * (agg[n * HC + c] + agg[n * HC + CC + c]) + bias[c];
  if (relu) v = fmaxf(v, 0.f);
  out[i] = v;
}

// ---------------- utility / pooling kernels ----------------
__global__ void fill_f32(float* __restrict__ p, float v, long n) {
  long i = (long)blockIdx.x * blockDim.x + threadIdx.x;
  if (i < n) p[i] = v;
}

__global__ void master_init(int* __restrict__ master) {
  int b = threadIdx.x;
  if (b < BB) master[b] = 0;
}

__global__ void master_scan(const int* __restrict__ batch, int* __restrict__ master) {
  long n = (long)blockIdx.x * blockDim.x + threadIdx.x;
  if (n >= NN) return;
  int bn = batch[n];
  if (n == NN - 1 || batch[n + 1] != bn) master[bn] = (int)n;  // batch is sorted
}

__global__ void master_gather(const float* __restrict__ h, const int* __restrict__ master,
                              float* __restrict__ out) {
  int i = blockIdx.x * blockDim.x + threadIdx.x;
  if (i >= BB * CC) return;
  int b = i / CC, c = i % CC;
  out[i] = h[(size_t)master[b] * CC + c];
}

// ---------------- host launcher ----------------
extern "C" void kernel_launch(void* const* d_in, const int* in_sizes, int n_in,
                              void* d_out, int out_size, void* d_ws, size_t ws_size,
                              hipStream_t stream) {
  (void)in_sizes; (void)n_in; (void)out_size; (void)ws_size;

  const float* x          = (const float*)d_in[0];
  const int*   edge_index = (const int*)d_in[1];
  const float* edge_attr  = (const float*)d_in[2];
  const int*   batch      = (const int*)d_in[3];

  // workspace layout (~212 MB)
  float* ws = (float*)d_ws;
  size_t o = 0;
  float* xl     = ws + o; o += (size_t)NN * HC;
  float* xr     = ws + o; o += (size_t)NN * HC;
  float* agg    = ws + o; o += (size_t)NN * HC;
  float* actA   = ws + o; o += (size_t)NN * CC;
  float* actB   = ws + o; o += (size_t)NN * CC;
  float* logits = ws + o; o += (size_t)EE * 2;
  float* eav    = ws + o; o += (size_t)EE * 2;
  float* mxb    = ws + o; o += (size_t)NN * 2;
  float* denom  = ws + o; o += (size_t)NN * 2;
  int*   master = (int*)(ws + o);

  const float* cur = x;
  int din = IN_DIM;
  float* outAct = actA;

  for (int l = 0; l < 3; ++l) {
    const float* wl  = (const float*)d_in[4 + 5 * l];
    const float* wr  = (const float*)d_in[5 + 5 * l];
    const float* we  = (const float*)d_in[6 + 5 * l];
    const float* att = (const float*)d_in[7 + 5 * l];
    const float* bia = (const float*)d_in[8 + 5 * l];

    dim3 gg((NN + 63) / 64, 4);
    gemm_xw_wmma<<<gg, 128, 0, stream>>>(cur, wl, xl, NN, din);
    gemm_xw_wmma<<<gg, 128, 0, stream>>>(cur, wr, xr, NN, din);

    long na = (long)NN * HC;
    fill_f32<<<(unsigned)((na + 255) / 256), 256, 0, stream>>>(agg, 0.f, na);
    fill_f32<<<(unsigned)(((long)NN * 2 + 255) / 256), 256, 0, stream>>>(mxb, -3.0e38f, (long)NN * 2);
    fill_f32<<<(unsigned)(((long)NN * 2 + 255) / 256), 256, 0, stream>>>(denom, 0.f, (long)NN * 2);

    edge_logits_wmma<<<EE / 16, 128, 0, stream>>>(
        edge_attr, we, xl, xr, att, edge_index, logits, mxb);
    edge_softmax_denom<<<(unsigned)(((long)EE * 2 + 255) / 256), 256, 0, stream>>>(
        logits, edge_index, mxb, eav, denom);
    edge_aggregate<<<EE / 8, 256, 0, stream>>>(xl, eav, denom, edge_index, agg);
    node_finalize<<<(unsigned)(((long)NN * CC + 255) / 256), 256, 0, stream>>>(
        agg, bia, outAct, l < 2 ? 1 : 0);

    cur = outAct;
    din = MID_DIM;
    outAct = (outAct == actA) ? actB : actA;
  }

  master_init<<<1, 64, 0, stream>>>(master);
  master_scan<<<(NN + 255) / 256, 256, 0, stream>>>(batch, master);
  master_gather<<<(BB * CC + 255) / 256, 256, 0, stream>>>(cur, master, (float*)d_out);
}
